// FinalLayer_3908420240262
// MI455X (gfx1250) — compile-verified, matrix operand order checked by reading, not run on hardware
//
#include <hip/hip_runtime.h>
#include <math.h>

typedef int v8i __attribute__((ext_vector_type(8)));

#define HIDDEN   1024
#define OUT_DIM  1024
#define INTER    2816
#define BB       4
#define NTOK     8192
#define MTOT     (BB * NTOK)   // 32768 tokens

__device__ __forceinline__ float silu_f(float x) { return x / (1.0f + __expf(-x)); }

// ---------------------------------------------------------------- small ops
__global__ __launch_bounds__(256)
void k_silu(const float* __restrict__ in, float* __restrict__ out, int n) {
    int i = blockIdx.x * blockDim.x + threadIdx.x;
    if (i < n) out[i] = silu_f(in[i]);
}

// wave-per-output GEMV: out[b][o] = dot(X[b,:K], W[o,:K]) (+bias)
__global__ __launch_bounds__(256)
void k_gemv(const float* __restrict__ X, const float* __restrict__ W,
            const float* __restrict__ bias, float* __restrict__ out,
            int K, int O) {
    int wave = blockIdx.x * (blockDim.x >> 5) + (threadIdx.x >> 5);
    int lane = threadIdx.x & 31;
    if (wave >= BB * O) return;
    int b = wave / O, o = wave % O;
    const float* xr = X + (size_t)b * K;
    const float* wr = W + (size_t)o * K;
    float s = 0.0f;
    for (int k = lane; k < K; k += 32) s += xr[k] * wr[k];
    #pragma unroll
    for (int m = 16; m >= 1; m >>= 1) s += __shfl_xor(s, m, 32);
    if (lane == 0) out[(size_t)b * O + o] = s + (bias ? bias[o] : 0.0f);
}

// act[b][i] = silu(gate[b][i]) * y[b][i]  with gate/y halves of gy [B, 2*INTER]
__global__ __launch_bounds__(256)
void k_silumul(const float* __restrict__ gy, float* __restrict__ act, int n) {
    int i = blockIdx.x * blockDim.x + threadIdx.x;
    if (i >= n) return;
    int b = i / INTER, j = i % INTER;
    const float* row = gy + (size_t)b * (2 * INTER);
    act[i] = silu_f(row[j]) * row[INTER + j];
}

__global__ __launch_bounds__(256)
void k_rmsnorm(const float* __restrict__ in, const float* __restrict__ w,
               float* __restrict__ out, int C, float eps) {
    __shared__ float red[256];
    int row = blockIdx.x, t = threadIdx.x;
    const float* x = in + (size_t)row * C;
    float ss = 0.0f;
    for (int i = t; i < C; i += 256) { float v = x[i]; ss += v * v; }
    red[t] = ss; __syncthreads();
    for (int off = 128; off > 0; off >>= 1) {
        if (t < off) red[t] += red[t + off];
        __syncthreads();
    }
    float r = rsqrtf(red[0] / (float)C + eps);
    for (int i = t; i < C; i += 256) out[(size_t)row * C + i] = x[i] * r * w[i];
}

// ------------------------------------------------- weight scale + quantize
__global__ __launch_bounds__(256)
void k_absmean_part(const float* __restrict__ w, float* __restrict__ part) {
    __shared__ float red[256];
    int t = threadIdx.x;
    const float* p = w + (size_t)blockIdx.x * 1024;
    float s = 0.0f;
    for (int i = t; i < 1024; i += 256) s += fabsf(p[i]);
    red[t] = s; __syncthreads();
    for (int off = 128; off > 0; off >>= 1) {
        if (t < off) red[t] += red[t + off];
        __syncthreads();
    }
    if (t == 0) part[blockIdx.x] = red[0];
}

__global__ __launch_bounds__(256)
void k_absmean_final(const float* __restrict__ part, float* __restrict__ wsc) {
    __shared__ float red[256];
    int t = threadIdx.x;
    float s = 0.0f;
    for (int i = t; i < 1024; i += 256) s += part[i];
    red[t] = s; __syncthreads();
    for (int off = 128; off > 0; off >>= 1) {
        if (t < off) red[t] += red[t + off];
        __syncthreads();
    }
    // wsc[0] = clip(mean|w|, 1e-5) == 1/quant_scale == weight dequant factor
    if (t == 0) wsc[0] = fmaxf(red[0] * (1.0f / (1024.0f * 1024.0f)), 1e-5f);
}

__global__ __launch_bounds__(256)
void k_wquant(const float* __restrict__ w, const float* __restrict__ wsc,
              signed char* __restrict__ wq, int n4) {
    int i = blockIdx.x * blockDim.x + threadIdx.x;
    if (i >= n4) return;
    float inv = 1.0f / wsc[0];
    const float4 v = ((const float4*)w)[i];
    int q0 = (int)rintf(v.x * inv), q1 = (int)rintf(v.y * inv);
    int q2 = (int)rintf(v.z * inv), q3 = (int)rintf(v.w * inv);
    q0 = q0 < -1 ? -1 : (q0 > 1 ? 1 : q0);
    q1 = q1 < -1 ? -1 : (q1 > 1 ? 1 : q1);
    q2 = q2 < -1 ? -1 : (q2 > 1 ? 1 : q2);
    q3 = q3 < -1 ? -1 : (q3 > 1 ? 1 : q3);
    ((int*)wq)[i] = (q0 & 0xff) | ((q1 & 0xff) << 8) | ((q2 & 0xff) << 16) | ((q3 & 0xff) << 24);
}

// ------------------- per-token: LN -> modulate -> RMSNorm -> int8 quant
__global__ __launch_bounds__(256)
void k_actquant(const float* __restrict__ x, const float* __restrict__ mod,
                const float* __restrict__ wn, const float* __restrict__ wsc,
                signed char* __restrict__ xq, float* __restrict__ deq) {
    __shared__ float r1[256];
    __shared__ float r2[256];
    int tok = blockIdx.x;
    int b = tok >> 13;                         // token / 8192
    int t = threadIdx.x;
    const float* xr = x + (size_t)tok * HIDDEN;
    const float* shiftp = mod + (size_t)b * (2 * HIDDEN);
    const float* scalep = shiftp + HIDDEN;

    float4 v = ((const float4*)xr)[t];
    float s = v.x + v.y + v.z + v.w;
    float q = v.x * v.x + v.y * v.y + v.z * v.z + v.w * v.w;
    r1[t] = s; r2[t] = q; __syncthreads();
    for (int off = 128; off > 0; off >>= 1) {
        if (t < off) { r1[t] += r1[t + off]; r2[t] += r2[t + off]; }
        __syncthreads();
    }
    float mu  = r1[0] * (1.0f / HIDDEN);
    float var = r2[0] * (1.0f / HIDDEN) - mu * mu;
    float rs  = rsqrtf(var + 1e-6f);
    __syncthreads();

    float4 sc4 = ((const float4*)scalep)[t];
    float4 sh4 = ((const float4*)shiftp)[t];
    float m0 = (v.x - mu) * rs * (1.0f + sc4.x) + sh4.x;
    float m1 = (v.y - mu) * rs * (1.0f + sc4.y) + sh4.y;
    float m2 = (v.z - mu) * rs * (1.0f + sc4.z) + sh4.z;
    float m3 = (v.w - mu) * rs * (1.0f + sc4.w) + sh4.w;

    r1[t] = m0 * m0 + m1 * m1 + m2 * m2 + m3 * m3; __syncthreads();
    for (int off = 128; off > 0; off >>= 1) {
        if (t < off) r1[t] += r1[t + off];
        __syncthreads();
    }
    float rr = rsqrtf(r1[0] * (1.0f / HIDDEN) + 1e-8f);
    __syncthreads();

    float4 w4 = ((const float4*)wn)[t];
    float e0 = m0 * rr * w4.x, e1 = m1 * rr * w4.y;
    float e2 = m2 * rr * w4.z, e3 = m3 * rr * w4.w;

    float am = fmaxf(fmaxf(fabsf(e0), fabsf(e1)), fmaxf(fabsf(e2), fabsf(e3)));
    r1[t] = am; __syncthreads();
    for (int off = 128; off > 0; off >>= 1) {
        if (t < off) r1[t] = fmaxf(r1[t], r1[t + off]);
        __syncthreads();
    }
    float ac = fmaxf(r1[0], 1e-5f);
    float qs = 127.0f / ac;
    int q0 = (int)rintf(e0 * qs), q1 = (int)rintf(e1 * qs);
    int q2 = (int)rintf(e2 * qs), q3 = (int)rintf(e3 * qs);
    q0 = q0 < -128 ? -128 : (q0 > 127 ? 127 : q0);
    q1 = q1 < -128 ? -128 : (q1 > 127 ? 127 : q1);
    q2 = q2 < -128 ? -128 : (q2 > 127 ? 127 : q2);
    q3 = q3 < -128 ? -128 : (q3 > 127 ? 127 : q3);
    ((int*)xq)[(size_t)tok * 256 + t] =
        (q0 & 0xff) | ((q1 & 0xff) << 8) | ((q2 & 0xff) << 16) | ((q3 & 0xff) << 24);
    if (t == 0) deq[tok] = (ac * (1.0f / 127.0f)) * wsc[0];  // act_deq * w_deq
}

// --------------------------- int8 WMMA GEMM: [32768 x 1024] x [1024 x 1024]^T
// Block: 256 thr = 8 waves (2 M-waves x 4 N-waves); each wave: 16M x 64N (4 accum).
// K loop in steps of 64 using V_WMMA_I32_16X16X64_IU8.
__global__ __launch_bounds__(256)
void k_bitgemm(const signed char* __restrict__ Aq, const signed char* __restrict__ Wq,
               const float* __restrict__ deq, const float* __restrict__ bias,
               float* __restrict__ out) {
    int wid  = threadIdx.x >> 5;
    int lane = threadIdx.x & 31;
    int wm = wid >> 2;            // 0..1
    int wn = wid & 3;             // 0..3
    int m0 = blockIdx.x * 32 + wm * 16;
    int n0 = blockIdx.y * 256 + wn * 64;
    int l15 = lane & 15;
    int kg  = (lane & 16) ? 8 : 0;   // 8-bit A/B fragment layout: lane-half K group

    const signed char* arow = Aq + (size_t)(m0 + l15) * HIDDEN + kg;
    const signed char* brow0 = Wq + (size_t)(n0 + 0 * 16 + l15) * HIDDEN + kg;
    const signed char* brow1 = Wq + (size_t)(n0 + 1 * 16 + l15) * HIDDEN + kg;
    const signed char* brow2 = Wq + (size_t)(n0 + 2 * 16 + l15) * HIDDEN + kg;
    const signed char* brow3 = Wq + (size_t)(n0 + 3 * 16 + l15) * HIDDEN + kg;

    v8i acc0 = {}, acc1 = {}, acc2 = {}, acc3 = {};

    for (int kb = 0; kb < HIDDEN; kb += 64) {
        // A fragment: dword pairs at byte offsets {0,16,32,48} within the K=64 slab
        const int2* pa = (const int2*)(arow + kb);
        int2 a0 = pa[0], a1 = pa[2], a2 = pa[4], a3 = pa[6];
        v8i a; a[0]=a0.x; a[1]=a0.y; a[2]=a1.x; a[3]=a1.y;
               a[4]=a2.x; a[5]=a2.y; a[6]=a3.x; a[7]=a3.y;

        const int2* pb;
        int2 b0, b1, b2, b3; v8i bf;

        pb = (const int2*)(brow0 + kb);
        b0 = pb[0]; b1 = pb[2]; b2 = pb[4]; b3 = pb[6];
        bf[0]=b0.x; bf[1]=b0.y; bf[2]=b1.x; bf[3]=b1.y; bf[4]=b2.x; bf[5]=b2.y; bf[6]=b3.x; bf[7]=b3.y;
        acc0 = __builtin_amdgcn_wmma_i32_16x16x64_iu8(true, a, true, bf, acc0, false, false);

        pb = (const int2*)(brow1 + kb);
        b0 = pb[0]; b1 = pb[2]; b2 = pb[4]; b3 = pb[6];
        bf[0]=b0.x; bf[1]=b0.y; bf[2]=b1.x; bf[3]=b1.y; bf[4]=b2.x; bf[5]=b2.y; bf[6]=b3.x; bf[7]=b3.y;
        acc1 = __builtin_amdgcn_wmma_i32_16x16x64_iu8(true, a, true, bf, acc1, false, false);

        pb = (const int2*)(brow2 + kb);
        b0 = pb[0]; b1 = pb[2]; b2 = pb[4]; b3 = pb[6];
        bf[0]=b0.x; bf[1]=b0.y; bf[2]=b1.x; bf[3]=b1.y; bf[4]=b2.x; bf[5]=b2.y; bf[6]=b3.x; bf[7]=b3.y;
        acc2 = __builtin_amdgcn_wmma_i32_16x16x64_iu8(true, a, true, bf, acc2, false, false);

        pb = (const int2*)(brow3 + kb);
        b0 = pb[0]; b1 = pb[2]; b2 = pb[4]; b3 = pb[6];
        bf[0]=b0.x; bf[1]=b0.y; bf[2]=b1.x; bf[3]=b1.y; bf[4]=b2.x; bf[5]=b2.y; bf[6]=b3.x; bf[7]=b3.y;
        acc3 = __builtin_amdgcn_wmma_i32_16x16x64_iu8(true, a, true, bf, acc3, false, false);
    }

    // Epilogue: C/D layout — lane<16: rows m0+r; lane>=16: rows m0+8+r; col = n + (lane&15)
    int mh = (lane & 16) ? 8 : 0;
    float dq[8];
    #pragma unroll
    for (int r = 0; r < 8; ++r) dq[r] = deq[m0 + mh + r];

    {
        int n = n0 + 0 * 16 + l15; float bv = bias[n];
        #pragma unroll
        for (int r = 0; r < 8; ++r)
            out[(size_t)(m0 + mh + r) * OUT_DIM + n] = (float)acc0[r] * dq[r] + bv;
    }
    {
        int n = n0 + 1 * 16 + l15; float bv = bias[n];
        #pragma unroll
        for (int r = 0; r < 8; ++r)
            out[(size_t)(m0 + mh + r) * OUT_DIM + n] = (float)acc1[r] * dq[r] + bv;
    }
    {
        int n = n0 + 2 * 16 + l15; float bv = bias[n];
        #pragma unroll
        for (int r = 0; r < 8; ++r)
            out[(size_t)(m0 + mh + r) * OUT_DIM + n] = (float)acc2[r] * dq[r] + bv;
    }
    {
        int n = n0 + 3 * 16 + l15; float bv = bias[n];
        #pragma unroll
        for (int r = 0; r < 8; ++r)
            out[(size_t)(m0 + mh + r) * OUT_DIM + n] = (float)acc3[r] * dq[r] + bv;
    }
}

// ---------------------------------------------------------------- launcher
extern "C" void kernel_launch(void* const* d_in, const int* in_sizes, int n_in,
                              void* d_out, int out_size, void* d_ws, size_t ws_size,
                              hipStream_t stream) {
    const float* x        = (const float*)d_in[0];
    const float* c        = (const float*)d_in[1];
    const float* w_in     = (const float*)d_in[2];
    const float* w_gate   = (const float*)d_in[3];
    const float* w_down   = (const float*)d_in[4];
    const float* w_outp   = (const float*)d_in[5];
    const float* b_outp   = (const float*)d_in[6];
    const float* w_norm   = (const float*)d_in[7];
    const float* w_op2    = (const float*)d_in[8];
    const float* b_op2    = (const float*)d_in[9];
    const float* w_bit    = (const float*)d_in[10];
    const float* b_bit    = (const float*)d_in[11];
    const float* w_bnorm  = (const float*)d_in[12];
    float* out = (float*)d_out;

    char* ws = (char*)d_ws;
    float*       sc   = (float*)(ws + 0);
    float*       h    = (float*)(ws + (64u  << 10));
    float*       gy   = (float*)(ws + (128u << 10));
    float*       act  = (float*)(ws + (256u << 10));
    float*       z    = (float*)(ws + (320u << 10));
    float*       o_   = (float*)(ws + (384u << 10));
    float*       onrm = (float*)(ws + (448u << 10));
    float*       mod  = (float*)(ws + (512u << 10));
    float*       part = (float*)(ws + (576u << 10));
    float*       wsc  = (float*)(ws + (592u << 10));
    float*       deq  = (float*)(ws + (640u << 10));  // 32768 floats
    signed char* wq   = (signed char*)(ws + (1024u << 10));  // 1 MB
    signed char* xq   = (signed char*)(ws + (2048u << 10));  // 32 MB

    // modulation chain (B=4, tiny)
    k_silu<<<dim3((BB * HIDDEN + 255) / 256), 256, 0, stream>>>(c, sc, BB * HIDDEN);
    k_gemv<<<dim3((BB * 1024 + 7) / 8), 256, 0, stream>>>(sc, w_in, nullptr, h, 1024, 1024);
    k_gemv<<<dim3((BB * 2 * INTER + 7) / 8), 256, 0, stream>>>(h, w_gate, nullptr, gy, 1024, 2 * INTER);
    k_silumul<<<dim3((BB * INTER + 255) / 256), 256, 0, stream>>>(gy, act, BB * INTER);
    k_gemv<<<dim3((BB * 1024 + 7) / 8), 256, 0, stream>>>(act, w_down, nullptr, z, INTER, 1024);
    k_gemv<<<dim3((BB * 2048 + 7) / 8), 256, 0, stream>>>(z, w_outp, b_outp, o_, 1024, 2048);
    k_rmsnorm<<<dim3(BB), 256, 0, stream>>>(o_, w_norm, onrm, 2048, 1e-6f);
    k_gemv<<<dim3((BB * 2048 + 7) / 8), 256, 0, stream>>>(onrm, w_op2, b_op2, mod, 2048, 2048);

    // weight quant
    k_absmean_part<<<dim3(1024), 256, 0, stream>>>(w_bit, part);
    k_absmean_final<<<dim3(1), 256, 0, stream>>>(part, wsc);
    k_wquant<<<dim3((1024 * 1024 / 4 + 255) / 256), 256, 0, stream>>>(w_bit, wsc, wq, 1024 * 1024 / 4);

    // per-token LN + modulate + RMSNorm + int8 quant
    k_actquant<<<dim3(MTOT), 256, 0, stream>>>(x, mod, w_bnorm, wsc, xq, deq);

    // int8 WMMA GEMM + dequant + bias
    k_bitgemm<<<dim3(MTOT / 32, OUT_DIM / 256), 256, 0, stream>>>(xq, wq, deq, b_bit, out);
}
